// Attention_64123861729812
// MI455X (gfx1250) — matrix operand order repaired and checked by reading, not compiled, not run on hardware
//
#include <hip/hip_runtime.h>
#include <hip/hip_bf16.h>

// ---------------------------------------------------------------------------
// GPT-2 attention block for MI455X (gfx1250).
// bf16 WMMA + fp32 softmax, TDM-staged weights, software-pipelined GEMMs.
// B=4, T=2048, E=1024, H=16, D=64.
// ---------------------------------------------------------------------------

typedef __attribute__((ext_vector_type(16))) __bf16        v16bf;
typedef __attribute__((ext_vector_type(8)))  float         v8f;
typedef __attribute__((ext_vector_type(8)))  unsigned int  v8u;
typedef __attribute__((ext_vector_type(4)))  unsigned int  v4u;
typedef __attribute__((ext_vector_type(8)))  int           v8i;
typedef __attribute__((ext_vector_type(4)))  int           v4i;

#define BB 4
#define TT 2048
#define EE 1024
#define HH 16
#define DD 64
#define ROWS (BB * TT)      // 8192
#define E3   (3 * EE)       // 3072

// fp32 -> bf16 round-to-nearest-even
__device__ __forceinline__ unsigned short f2bf(float f) {
  unsigned int u = __builtin_bit_cast(unsigned int, f);
  u += 0x7fffu + ((u >> 16) & 1u);
  return (unsigned short)(u >> 16);
}

__device__ __forceinline__ v16bf bf_frag_v(uint4 lo, uint4 hi) {
  v8u u;
  u[0] = lo.x; u[1] = lo.y; u[2] = lo.z; u[3] = lo.w;
  u[4] = hi.x; u[5] = hi.y; u[6] = hi.z; u[7] = hi.w;
  return __builtin_bit_cast(v16bf, u);
}

__device__ __forceinline__ v16bf bf_frag(const unsigned int* pa,
                                         const unsigned int* pb) {
  return bf_frag_v(*(const uint4*)pa, *(const uint4*)pb);
}

__device__ __forceinline__ v8f wmma_bf16(v16bf a, v16bf b, v8f c) {
  return __builtin_amdgcn_wmma_f32_16x16x32_bf16(false, a, false, b,
                                                 (short)0, c, false, false);
}

// ---------------------------------------------------------------------------
// Tensor Data Mover: 2-D tile Global -> LDS.
//   rows x (tile_bytes_x) bytes per row, global row stride = stride_bytes.
//   data_size = 8B units; tile start must be 8B aligned.
// ---------------------------------------------------------------------------
__device__ __forceinline__ void tdm_load_2d(unsigned lds_off, const void* gptr,
                                            unsigned tile_bytes_x,
                                            unsigned rows,
                                            unsigned stride_bytes) {
  unsigned long long ga = (unsigned long long)(__SIZE_TYPE__)gptr;
  unsigned ux = tile_bytes_x >> 3;     // units of 8B
  unsigned s0 = stride_bytes >> 3;     // units of 8B
  v4u g0;
  g0[0] = 1u;                                           // count=1, user mode
  g0[1] = lds_off;                                      // lds_addr (bytes)
  g0[2] = (unsigned)(ga & 0xffffffffu);                 // global_addr lo
  g0[3] = (unsigned)((ga >> 32) & 0x01ffffffu) | (2u << 30); // addr hi | type=2
  v8i g1;
  g1[0] = (int)(3u << 16);                 // data_size=8B, wg_mask=0, flags=0
  g1[1] = (int)(ux << 16);                 // tensor_dim0[15:0]
  g1[2] = (int)((ux >> 16) | (rows << 16)); // tensor_dim0 hi | tensor_dim1 lo
  g1[3] = (int)((rows >> 16) | (ux << 16)); // tensor_dim1 hi | tile_dim0
  g1[4] = (int)rows;                       // tile_dim1 (tile_dim2=0)
  g1[5] = (int)s0;                         // tensor_dim0_stride lo
  g1[6] = 0;                               // stride0 hi | stride1 lo
  g1[7] = 0;
  v4i z4 = {0, 0, 0, 0};
#if defined(__clang_major__) && (__clang_major__ >= 23)
  v8i z8 = {0, 0, 0, 0, 0, 0, 0, 0};
  __builtin_amdgcn_tensor_load_to_lds(g0, g1, z4, z4, z8, 0);
#else
  __builtin_amdgcn_tensor_load_to_lds(g0, g1, z4, z4, 0);
#endif
}

// ---------------------------------------------------------------------------
// GEMM building blocks (wave tile 32x64, K-step 32)
// ---------------------------------------------------------------------------
struct AFrag { uint4 q00, q01, q10, q11; };

__device__ __forceinline__ AFrag load_a(const unsigned int* arow0,
                                        const unsigned int* arow1,
                                        int kk, int half) {
  AFrag a;
  const unsigned int* p0 = arow0 + kk * 16;
  const unsigned int* p1 = arow1 + kk * 16;
  a.q00 = *(const uint4*)(p0 + half * 4);
  a.q01 = *(const uint4*)(p0 + 8 + half * 4);
  a.q10 = *(const uint4*)(p1 + half * 4);
  a.q11 = *(const uint4*)(p1 + 8 + half * 4);
  return a;
}

// Issue all 8 ds_load_b128 first (sched barrier keeps them hoisted -> slack
// DS waits), then run the 8 WMMAs.
__device__ __forceinline__ void gemm_step(const unsigned char* bbuf,
                                          int ml, int half, const AFrag& a,
                                          v8f c0[4], v8f c1[4]) {
  v16bf A0 = bf_frag_v(a.q00, a.q01);
  v16bf A1 = bf_frag_v(a.q10, a.q11);
  v16bf Bf[4];
#pragma unroll
  for (int dt = 0; dt < 4; ++dt) {
    const unsigned int* bp =
        (const unsigned int*)(bbuf + (dt * 16 + ml) * 64 + half * 16);
    Bf[dt] = bf_frag(bp, bp + 8);
  }
  __builtin_amdgcn_sched_barrier(0);   // do not sink the B loads into the MMAs
#pragma unroll
  for (int dt = 0; dt < 4; ++dt) {
    c0[dt] = wmma_bf16(A0, Bf[dt], c0[dt]);
    c1[dt] = wmma_bf16(A1, Bf[dt], c1[dt]);
  }
}

// ---------------------------------------------------------------------------
// Stage 0: precision conversion / weight transposition (one-time, L2-bound)
// ---------------------------------------------------------------------------
__global__ void __launch_bounds__(256) cvt_bf16_kernel(
    const float* __restrict__ in, unsigned short* __restrict__ out, int n) {
  int i = blockIdx.x * 256 + threadIdx.x;
  if (i < n) out[i] = f2bf(in[i]);
}

// out[n*K + k] = bf16(in[k*N + n])   (in: [K,N] fp32 row-major)
__global__ void __launch_bounds__(256) transpose_bf16_kernel(
    const float* __restrict__ in, unsigned short* __restrict__ out,
    int K, int N) {
  int i = blockIdx.x * 256 + threadIdx.x;
  if (i < K * N) {
    int n = i / K;
    int k = i - n * K;
    out[i] = f2bf(in[k * N + n]);
  }
}

// ---------------------------------------------------------------------------
// Stage 1: QKV GEMM.  qkv = x @ W_attn + b_attn.
// Wave: 32 rows x 64 cols.  K loop unrolled x2 with ping-pong A registers
// and ping-pong TDM LDS buffers (no register copies, slack waits).
// Q,K stored [B,H,T,D] bf16; V stored [B,H,D,T] bf16.
// ---------------------------------------------------------------------------
__global__ void __launch_bounds__(128) qkv_gemm_kernel(
    const unsigned int* __restrict__ xbf,   // [8192][512] dwords (bf16 pairs)
    const unsigned short* __restrict__ wt,  // [3072][1024] bf16 (W_attn^T)
    const float* __restrict__ b_attn,
    unsigned short* __restrict__ qws,
    unsigned short* __restrict__ kws,
    unsigned short* __restrict__ vws) {
  __shared__ __align__(16) unsigned char btile[4][2][4096];

  const int lane = threadIdx.x & 31;
  const int wave = threadIdx.x >> 5;
  const int m0   = blockIdx.x * 32;
  const int n0   = (blockIdx.y * 4 + wave) * 64;
  const int ml   = lane & 15;
  const int half = lane >> 4;

  const unsigned int* arow0 = xbf + (size_t)(m0 + ml) * 512;
  const unsigned int* arow1 = xbf + (size_t)(m0 + 16 + ml) * 512;
  const char* wtile = (const char*)(wt + (size_t)n0 * EE);

  unsigned lds0 = (unsigned)(__SIZE_TYPE__)(const void*)&btile[wave][0][0];
  unsigned lds1 = (unsigned)(__SIZE_TYPE__)(const void*)&btile[wave][1][0];
  const unsigned char* buf0 = &btile[wave][0][0];
  const unsigned char* buf1 = &btile[wave][1][0];

  v8f z = {0.f, 0.f, 0.f, 0.f, 0.f, 0.f, 0.f, 0.f};
  v8f c0[4], c1[4];
#pragma unroll
  for (int dt = 0; dt < 4; ++dt) { c0[dt] = z; c1[dt] = z; }

  // Prologue: TDM step 0 -> buf0, A step 0 -> ra.
  tdm_load_2d(lds0, wtile, 64, 64, EE * 2);
  AFrag ra = load_a(arow0, arow1, 0, half);
  AFrag rb;

  for (int kk = 0; kk < 32; kk += 2) {
    // ---- even phase: compute buf0/ra; prefetch kk+1 -> buf1/rb ------------
    tdm_load_2d(lds1, wtile + (kk + 1) * 64, 64, 64, EE * 2);
    rb = load_a(arow0, arow1, kk + 1, half);
    __builtin_amdgcn_s_wait_tensorcnt(1);
    __asm__ volatile("" ::: "memory");
    gemm_step(buf0, ml, half, ra, c0, c1);

    // ---- odd phase: compute buf1/rb; prefetch kk+2 -> buf0/ra -------------
    if (kk + 2 < 32) {
      tdm_load_2d(lds0, wtile + (kk + 2) * 64, 64, 64, EE * 2);
      ra = load_a(arow0, arow1, kk + 2, half);
      __builtin_amdgcn_s_wait_tensorcnt(1);
    } else {
      __builtin_amdgcn_s_wait_tensorcnt(0);
    }
    __asm__ volatile("" ::: "memory");
    gemm_step(buf1, ml, half, rb, c0, c1);
  }

  // Epilogue: scatter into Q/K/V layouts, bf16.
#pragma unroll
  for (int dt = 0; dt < 4; ++dt) {
    int col   = n0 + dt * 16 + ml;                // 0..3071
    int which = col >> 10;                        // 0=q 1=k 2=v
    int e     = col & (EE - 1);
    int h     = e >> 6;
    int d     = e & 63;
    float bias = b_attn[col];
#pragma unroll
    for (int s = 0; s < 2; ++s) {
#pragma unroll
      for (int r = 0; r < 8; ++r) {
        int row = m0 + s * 16 + r + 8 * half;     // 0..8191
        int b   = row >> 11;
        int t   = row & (TT - 1);
        int bh  = b * HH + h;
        float v = (s == 0 ? c0[dt][r] : c1[dt][r]) + bias;
        unsigned short bv = f2bf(v);
        if (which == 0)      qws[((size_t)bh * TT + t) * DD + d] = bv;
        else if (which == 1) kws[((size_t)bh * TT + t) * DD + d] = bv;
        else                 vws[((size_t)bh * DD + d) * TT + t] = bv;
      }
    }
  }
}

// ---------------------------------------------------------------------------
// Stage 2: flash attention.  One wave per 16-query tile, 32-key steps.
// V loads issued before score WMMAs (in flight during softmax + LDS hop).
// ---------------------------------------------------------------------------
__global__ void __launch_bounds__(128) flash_attn_kernel(
    const unsigned short* __restrict__ qws,   // [B*H][T][D] bf16
    const unsigned short* __restrict__ kws,   // [B*H][T][D] bf16
    const unsigned short* __restrict__ vws,   // [B*H][D][T] bf16
    unsigned short* __restrict__ aws) {       // [8192][1024] bf16
  __shared__ __align__(16) unsigned short sh[4][16 * 32];

  const int lane = threadIdx.x & 31;
  const int wave = threadIdx.x >> 5;
  const int qt   = blockIdx.x * 4 + wave;     // 0..127
  const int bh   = blockIdx.y;                // 0..63
  const int b    = bh >> 4;
  const int h    = bh & 15;
  const int q0   = qt * 16;
  const int ml   = lane & 15;
  const int half = lane >> 4;

  const unsigned int* qrow =
      (const unsigned int*)qws + ((size_t)bh * TT + q0 + ml) * 32;
  v16bf qa0 = bf_frag(qrow + half * 8,      qrow + half * 8 + 4);
  v16bf qa1 = bf_frag(qrow + 16 + half * 8, qrow + 16 + half * 8 + 4);

  v8f z = {0.f, 0.f, 0.f, 0.f, 0.f, 0.f, 0.f, 0.f};
  v8f o[4];
  o[0] = z; o[1] = z; o[2] = z; o[3] = z;

  float mstate[8], lstate[8];
#pragma unroll
  for (int r = 0; r < 8; ++r) { mstate[r] = -3.0e38f; lstate[r] = 0.f; }

  unsigned short* shp = sh[wave];

  for (int k0 = 0; k0 < q0 + 16; k0 += 32) {
    // ---- K fragments (needed first by the score WMMAs) --------------------
    const unsigned int* kr0 =
        (const unsigned int*)kws + ((size_t)bh * TT + k0 + ml) * 32;
    const unsigned int* kr1 =
        (const unsigned int*)kws + ((size_t)bh * TT + k0 + 16 + ml) * 32;
    v16bf kb00 = bf_frag(kr0 + half * 8,      kr0 + half * 8 + 4);
    v16bf kb01 = bf_frag(kr0 + 16 + half * 8, kr0 + 16 + half * 8 + 4);
    v16bf kb10 = bf_frag(kr1 + half * 8,      kr1 + half * 8 + 4);
    v16bf kb11 = bf_frag(kr1 + 16 + half * 8, kr1 + 16 + half * 8 + 4);

    // ---- issue V loads now: they complete during softmax ------------------
    uint4 vlo[4], vhi[4];
#pragma unroll
    for (int dt = 0; dt < 4; ++dt) {
      const unsigned int* vr = (const unsigned int*)vws +
          ((size_t)bh * DD + dt * 16 + ml) * (TT / 2) + (k0 >> 1) + half * 8;
      vlo[dt] = *(const uint4*)vr;
      vhi[dt] = *(const uint4*)(vr + 4);
    }

    // ---- prefetch next key step's K rows ----------------------------------
    if (k0 + 32 < q0 + 16) {
      __builtin_prefetch((const void*)(kr0 + 1024), 0, 3);
      __builtin_prefetch((const void*)(kr1 + 1024), 0, 3);
    }

    v8f s0 = z, s1 = z;
    s0 = wmma_bf16(qa0, kb00, s0);
    s0 = wmma_bf16(qa1, kb01, s0);
    s1 = wmma_bf16(qa0, kb10, s1);
    s1 = wmma_bf16(qa1, kb11, s1);

    // ---- online softmax bookkeeping (fp32) --------------------------------
#pragma unroll
    for (int r = 0; r < 8; ++r) {
      int qg = q0 + r + 8 * half;
      float x0 = s0[r] * 0.125f;                // 1/sqrt(64)
      float x1 = s1[r] * 0.125f;
      if (k0 + ml > qg)      x0 = -1.0e30f;     // causal mask
      if (k0 + 16 + ml > qg) x1 = -1.0e30f;

      float rm = fmaxf(x0, x1);
      rm = fmaxf(rm, __shfl_xor(rm, 1));
      rm = fmaxf(rm, __shfl_xor(rm, 2));
      rm = fmaxf(rm, __shfl_xor(rm, 4));
      rm = fmaxf(rm, __shfl_xor(rm, 8));

      float mnew  = fmaxf(mstate[r], rm);
      float alpha = __expf(mstate[r] - mnew);
      mstate[r]   = mnew;

      float p0 = __expf(x0 - mnew);
      float p1 = __expf(x1 - mnew);
      float ps = p0 + p1;
      ps += __shfl_xor(ps, 1);
      ps += __shfl_xor(ps, 2);
      ps += __shfl_xor(ps, 4);
      ps += __shfl_xor(ps, 8);
      lstate[r] = lstate[r] * alpha + ps;

      o[0][r] *= alpha; o[1][r] *= alpha; o[2][r] *= alpha; o[3][r] *= alpha;

      int mr = r + 8 * half;
      shp[mr * 32 + ml]      = f2bf(p0);        // same-wave DS: in-order
      shp[mr * 32 + 16 + ml] = f2bf(p1);
    }

    // ---- re-layout P (C tile -> A fragment) via LDS -----------------------
    const unsigned int* pr = (const unsigned int*)shp + ml * 16;
    v16bf pa = bf_frag(pr + half * 8, pr + half * 8 + 4);

    // ---- O += P(16x32) @ V(32x64) -----------------------------------------
#pragma unroll
    for (int dt = 0; dt < 4; ++dt) {
      v16bf vb = bf_frag_v(vlo[dt], vhi[dt]);
      o[dt] = wmma_bf16(pa, vb, o[dt]);
    }
  }

  // ---- normalize and store attention output [8192,1024] bf16 --------------
#pragma unroll
  for (int r = 0; r < 8; ++r) {
    float inv = 1.0f / lstate[r];
    int row = b * TT + q0 + r + 8 * half;
#pragma unroll
    for (int dt = 0; dt < 4; ++dt) {
      int col = h * DD + dt * 16 + ml;
      aws[(size_t)row * EE + col] = f2bf(o[dt][r] * inv);
    }
  }
}

// ---------------------------------------------------------------------------
// Stage 3: projection GEMM.  y = attn @ W_proj + b_proj -> fp32.
// Same skeleton as stage 1.
// ---------------------------------------------------------------------------
__global__ void __launch_bounds__(128) proj_gemm_kernel(
    const unsigned int* __restrict__ abf,    // [8192][512] dwords (bf16)
    const unsigned short* __restrict__ wpt,  // [1024][1024] bf16 (W_proj^T)
    const float* __restrict__ b_proj,
    float* __restrict__ out) {
  __shared__ __align__(16) unsigned char btile[4][2][4096];

  const int lane = threadIdx.x & 31;
  const int wave = threadIdx.x >> 5;
  const int m0   = blockIdx.x * 32;
  const int n0   = (blockIdx.y * 4 + wave) * 64;
  const int ml   = lane & 15;
  const int half = lane >> 4;

  const unsigned int* arow0 = abf + (size_t)(m0 + ml) * 512;
  const unsigned int* arow1 = abf + (size_t)(m0 + 16 + ml) * 512;
  const char* wtile = (const char*)(wpt + (size_t)n0 * EE);

  unsigned lds0 = (unsigned)(__SIZE_TYPE__)(const void*)&btile[wave][0][0];
  unsigned lds1 = (unsigned)(__SIZE_TYPE__)(const void*)&btile[wave][1][0];
  const unsigned char* buf0 = &btile[wave][0][0];
  const unsigned char* buf1 = &btile[wave][1][0];

  v8f z = {0.f, 0.f, 0.f, 0.f, 0.f, 0.f, 0.f, 0.f};
  v8f c0[4], c1[4];
#pragma unroll
  for (int dt = 0; dt < 4; ++dt) { c0[dt] = z; c1[dt] = z; }

  tdm_load_2d(lds0, wtile, 64, 64, EE * 2);
  AFrag ra = load_a(arow0, arow1, 0, half);
  AFrag rb;

  for (int kk = 0; kk < 32; kk += 2) {
    tdm_load_2d(lds1, wtile + (kk + 1) * 64, 64, 64, EE * 2);
    rb = load_a(arow0, arow1, kk + 1, half);
    __builtin_amdgcn_s_wait_tensorcnt(1);
    __asm__ volatile("" ::: "memory");
    gemm_step(buf0, ml, half, ra, c0, c1);

    if (kk + 2 < 32) {
      tdm_load_2d(lds0, wtile + (kk + 2) * 64, 64, 64, EE * 2);
      ra = load_a(arow0, arow1, kk + 2, half);
      __builtin_amdgcn_s_wait_tensorcnt(1);
    } else {
      __builtin_amdgcn_s_wait_tensorcnt(0);
    }
    __asm__ volatile("" ::: "memory");
    gemm_step(buf1, ml, half, rb, c0, c1);
  }

#pragma unroll
  for (int dt = 0; dt < 4; ++dt) {
    int col = n0 + dt * 16 + ml;
    float bias = b_proj[col];
#pragma unroll
    for (int s = 0; s < 2; ++s) {
#pragma unroll
      for (int r = 0; r < 8; ++r) {
        int row = m0 + s * 16 + r + 8 * half;
        out[(size_t)row * EE + col] = (s == 0 ? c0[dt][r] : c1[dt][r]) + bias;
      }
    }
  }
}

// ---------------------------------------------------------------------------
// Host launcher
// ---------------------------------------------------------------------------
extern "C" void kernel_launch(void* const* d_in, const int* in_sizes, int n_in,
                              void* d_out, int out_size, void* d_ws,
                              size_t ws_size, hipStream_t stream) {
  (void)in_sizes; (void)n_in; (void)out_size; (void)ws_size;

  const float* x      = (const float*)d_in[0];
  /* d_in[1] = mask: causal mask handled analytically */
  const float* W_attn = (const float*)d_in[2];
  const float* b_attn = (const float*)d_in[3];
  const float* W_proj = (const float*)d_in[4];
  const float* b_proj = (const float*)d_in[5];
  float* out = (float*)d_out;

  // Workspace carve-up (bf16 buffers), ~88 MB total.
  char* ws = (char*)d_ws;
  unsigned short* xbf = (unsigned short*)ws;  ws += (size_t)ROWS * EE * 2;
  unsigned short* wt  = (unsigned short*)ws;  ws += (size_t)E3 * EE * 2;
  unsigned short* wpt = (unsigned short*)ws;  ws += (size_t)EE * EE * 2;
  unsigned short* qws = (unsigned short*)ws;  ws += (size_t)BB * HH * TT * DD * 2;
  unsigned short* kws = (unsigned short*)ws;  ws += (size_t)BB * HH * TT * DD * 2;
  unsigned short* vws = (unsigned short*)ws;  ws += (size_t)BB * HH * TT * DD * 2;
  unsigned short* aws = (unsigned short*)ws;

  // Stage 0: bf16 conversion + weight transposition
  int nx = ROWS * EE;
  cvt_bf16_kernel<<<(nx + 255) / 256, 256, 0, stream>>>(x, xbf, nx);
  int nwa = EE * E3;
  transpose_bf16_kernel<<<(nwa + 255) / 256, 256, 0, stream>>>(W_attn, wt, EE, E3);
  int nwp = EE * EE;
  transpose_bf16_kernel<<<(nwp + 255) / 256, 256, 0, stream>>>(W_proj, wpt, EE, EE);

  // Stage 1: QKV GEMM  (256 row-tiles of 32 x 12 col-strips of 256)
  qkv_gemm_kernel<<<dim3(ROWS / 32, E3 / 256), 128, 0, stream>>>(
      (const unsigned int*)xbf, wt, b_attn, qws, kws, vws);

  // Stage 2: flash attention  (128 q-tiles / 4 per block, 64 (b,h) pairs)
  flash_attn_kernel<<<dim3(TT / 16 / 4, BB * HH), 128, 0, stream>>>(
      qws, kws, vws, aws);

  // Stage 3: projection GEMM
  proj_gemm_kernel<<<dim3(ROWS / 32, EE / 256), 128, 0, stream>>>(
      (const unsigned int*)aws, wpt, b_proj, out);
}
